// Attention_21698174779774
// MI455X (gfx1250) — matrix operand order repaired and checked by reading, not compiled, not run on hardware
//
#include <hip/hip_runtime.h>

typedef _Float16 h8v  __attribute__((ext_vector_type(8)));
typedef _Float16 h16v __attribute__((ext_vector_type(16)));
typedef float    f4v  __attribute__((ext_vector_type(4)));
typedef float    f8v  __attribute__((ext_vector_type(8)));

#define DM    768
#define S_LEN 1024
#define NH    12
#define HD    64
#define KCH   24              // 768 / 32
#define WPERM 589824          // 768*768 halves per weight

__device__ __forceinline__ f8v wmma_f16(h16v a, h16v b, f8v c) {
  return __builtin_amdgcn_wmma_f32_16x16x32_f16(false, a, false, b, (short)0, c,
                                                false, false);
}

// ---------------------------------------------------------------------------
// Kernel 0: convert W{q,k,v,o} f32[768][768] into WMMA B-matrix swizzled f16.
// Layout: [w][nt(48)][kc(24)][lane(32)][j(16)], element = W[k][n],
//   k = kc*32 + (lane&16 ? 8 : 0) + (j&7) + (j&8 ? 16 : 0), n = nt*16 + (lane&15).
// ---------------------------------------------------------------------------
__global__ void k_prep(const float* __restrict__ Wq, const float* __restrict__ Wk,
                       const float* __restrict__ Wv, const float* __restrict__ Wo,
                       _Float16* __restrict__ wsw) {
  int tid = blockIdx.x * 256 + threadIdx.x;
  int j    = tid & 15;
  int lane = (tid >> 4) & 31;
  int r    = tid >> 9;
  int kc = r % KCH;  r /= KCH;
  int nt = r % 48;   r /= 48;
  const float* W = (r == 0) ? Wq : (r == 1) ? Wk : (r == 2) ? Wv : Wo;
  int k = kc * 32 + ((lane & 16) ? 8 : 0) + (j & 7) + ((j & 8) ? 16 : 0);
  int n = nt * 16 + (lane & 15);
  wsw[tid] = (_Float16)W[k * DM + n];
}

// ---------------------------------------------------------------------------
// Kernel 1: fused QKV projection + bias + RoPE.
// One wave computes a 16(row) x 64(one head) tile. K-loop 24 x (1 A-load + 4 WMMA).
// Q/K: RoPE then LDS-transpose -> row-major f16 [bh][s][64].
// V: stored transposed f16 [bh][d][s] directly (8 consecutive keys per lane).
// ---------------------------------------------------------------------------
__global__ void __launch_bounds__(128) k_qkv(
    const float* __restrict__ x,
    const float* __restrict__ bq, const float* __restrict__ bk,
    const float* __restrict__ bv,
    const _Float16* __restrict__ wsw,
    _Float16* __restrict__ qbuf, _Float16* __restrict__ kbuf,
    _Float16* __restrict__ vt) {
  __shared__ __align__(32) _Float16 lbuf[4][16 * 64];
  int widx = threadIdx.x >> 5;
  int lane = threadIdx.x & 31;
  int wv   = blockIdx.x * 4 + widx;
  int qkv  = wv / 12288;                 // 0=Q,1=K,2=V (uniform per block)
  int rem  = wv - qkv * 12288;
  int h    = rem >> 10;                  // head (uniform per block)
  int rt   = rem & 1023;                 // 16-row tile id over 16384 rows
  int n    = lane & 15;
  int hi   = (lane >> 4) & 1;
  int hi8  = hi * 8;
  int br   = rt >> 6;                    // batch*rep
  int seq0 = (rt & 63) << 4;             // sequence base of tile

  const float*    xrow = x + (size_t)(rt * 16 + n) * DM;
  const _Float16* wb   = wsw + (size_t)qkv * WPERM
                       + (size_t)(h * 4) * KCH * 512 + lane * 16;

  f8v zero = {};
  f8v acc[4];
#pragma unroll
  for (int t = 0; t < 4; ++t) acc[t] = zero;

  for (int kc = 0; kc < KCH; ++kc) {
    int kb0 = kc * 32;
    // issue all loads first: 8 B-dwords4 + 4 A-dwords4, then convert A while
    // B-tile loads are in flight, then the 4 chained WMMAs.
    h16v B[4];
#pragma unroll
    for (int t = 0; t < 4; ++t)
      B[t] = *(const h16v*)(wb + (size_t)(t * KCH + kc) * 512);
    f4v a0 = *(const f4v*)(xrow + kb0 + hi8);
    f4v a1 = *(const f4v*)(xrow + kb0 + hi8 + 4);
    f4v a2 = *(const f4v*)(xrow + kb0 + 16 + hi8);
    f4v a3 = *(const f4v*)(xrow + kb0 + 16 + hi8 + 4);
    h16v A;
#pragma unroll
    for (int e = 0; e < 4; ++e) {
      A[e]      = (_Float16)a0[e];
      A[4 + e]  = (_Float16)a1[e];
      A[8 + e]  = (_Float16)a2[e];
      A[12 + e] = (_Float16)a3[e];
    }
#pragma unroll
    for (int t = 0; t < 4; ++t) acc[t] = wmma_f16(A, B[t], acc[t]);
  }

  const float* bias = (qkv == 0) ? bq : (qkv == 1) ? bk : bv;
#pragma unroll
  for (int t = 0; t < 4; ++t) {
    float bvv = bias[h * HD + t * 16 + n];
#pragma unroll
    for (int i = 0; i < 8; ++i) acc[t][i] += bvv;
  }

  if (qkv < 2) {  // RoPE: pair (d, d+32) == accumulator tiles (t, t+2), same lane
    const float LN_BASE_OVER_32 = 0.2878231366242682f;  // ln(10000)/32
    float inv0 = __expf(-(float)n * LN_BASE_OVER_32);
    float inv1 = __expf(-(float)(n + 16) * LN_BASE_OVER_32);
#pragma unroll
    for (int i = 0; i < 8; ++i) {
      float pos = (float)(seq0 + hi8 + i);
      float sn0, cs0, sn1, cs1;
      __sincosf(pos * inv0, &sn0, &cs0);
      __sincosf(pos * inv1, &sn1, &cs1);
      float a0 = acc[0][i], a1 = acc[1][i], a2 = acc[2][i], a3 = acc[3][i];
      acc[0][i] = a0 * cs0 - a2 * sn0;
      acc[2][i] = a2 * cs0 + a0 * sn0;
      acc[1][i] = a1 * cs1 - a3 * sn1;
      acc[3][i] = a3 * cs1 + a1 * sn1;
    }
  }

  int bh = br * NH + h;
  if (qkv == 2) {  // V transposed: [bh][d][key], 8 consecutive keys per lane
#pragma unroll
    for (int t = 0; t < 4; ++t) {
      h8v p;
#pragma unroll
      for (int i = 0; i < 8; ++i) p[i] = (_Float16)acc[t][i];
      *(h8v*)(vt + ((size_t)bh * HD + t * 16 + n) * S_LEN + seq0 + hi8) = p;
    }
  } else {  // Q/K row-major via LDS transpose
#pragma unroll
    for (int t = 0; t < 4; ++t)
#pragma unroll
      for (int i = 0; i < 8; ++i)
        lbuf[widx][(i + hi8) * 64 + t * 16 + n] = (_Float16)acc[t][i];
    __syncthreads();
    _Float16* dst = ((qkv == 0) ? qbuf : kbuf) + ((size_t)bh * S_LEN + seq0) * HD;
    int colb = hi * 32;
#pragma unroll
    for (int c = 0; c < 4; ++c) {
      h8v v = *(h8v*)&lbuf[widx][n * 64 + colb + c * 8];
      *(h8v*)(dst + (size_t)n * HD + colb + c * 8) = v;
    }
  }
}

// ---------------------------------------------------------------------------
// Kernel 2: flash attention. Block = 4 waves = 64 query rows of one (br,head).
// Per wave: 16 q-rows, O accum 16x64 in 4 f32 tiles, online softmax over
// 32-key blocks. 8 WMMAs per block (4 for S, 4 for PV).
// ---------------------------------------------------------------------------
__global__ void __launch_bounds__(128) k_attn(
    const _Float16* __restrict__ qbuf, const _Float16* __restrict__ kbuf,
    const _Float16* __restrict__ vt, const int* __restrict__ mask,
    _Float16* __restrict__ attnb) {
  __shared__ __align__(32) _Float16 lbuf[4][16 * 64];
  int widx = threadIdx.x >> 5;
  int lane = threadIdx.x & 31;
  int bh   = blockIdx.x >> 4;
  int qblk = blockIdx.x & 15;
  int q0   = qblk * 64 + widx * 16;
  int br   = bh / NH;
  int h    = bh - br * NH;
  int n    = lane & 15;
  int hi   = (lane >> 4) & 1;
  int hi8  = hi * 8;
  int dhalf = hi * 16;

  // Q A-tiles, pre-scaled by 1/sqrt(64) = 0.125 (exact in f16)
  const _Float16* qp = qbuf + ((size_t)bh * S_LEN + q0 + n) * HD;
  h16v A0, A1;
  {
    h8v lo0 = *(const h8v*)(qp + dhalf);
    h8v hh0 = *(const h8v*)(qp + dhalf + 8);
    h8v lo1 = *(const h8v*)(qp + 32 + dhalf);
    h8v hh1 = *(const h8v*)(qp + 32 + dhalf + 8);
#pragma unroll
    for (int e = 0; e < 8; ++e) {
      A0[e]     = lo0[e] * (_Float16)0.125f;
      A0[8 + e] = hh0[e] * (_Float16)0.125f;
      A1[e]     = lo1[e] * (_Float16)0.125f;
      A1[8 + e] = hh1[e] * (_Float16)0.125f;
    }
  }
  const _Float16* kp = kbuf + (size_t)bh * S_LEN * HD;
  const _Float16* vp = vt + (size_t)bh * HD * S_LEN;
  const int*      mp = mask + br * S_LEN;

  f8v zero = {};
  f8v o[4];
  float mrow[8], lrow[8];
#pragma unroll
  for (int t = 0; t < 4; ++t) o[t] = zero;
#pragma unroll
  for (int i = 0; i < 8; ++i) { mrow[i] = -3.0e38f; lrow[i] = 0.0f; }

  for (int kb = 0; kb < S_LEN; kb += 32) {
    if (kb + 32 < S_LEN) {  // speculative prefetch of next key/value block
      __builtin_prefetch(kp + (size_t)(kb + 32 + n) * HD, 0, 1);
      __builtin_prefetch(vp + (size_t)n * S_LEN + kb + 32, 0, 1);
    }
    int mk0 = mp[kb + n];
    int mk1 = mp[kb + 16 + n];

    // S = (Q/8) @ K^T : issue all 4 K B-tile loads, then 4 WMMAs
    const _Float16* kr0 = kp + (size_t)(kb + n) * HD + dhalf;
    const _Float16* kr1 = kp + (size_t)(kb + 16 + n) * HD + dhalf;
    h16v b00 = *(const h16v*)(kr0);
    h16v b01 = *(const h16v*)(kr0 + 32);
    h16v b10 = *(const h16v*)(kr1);
    h16v b11 = *(const h16v*)(kr1 + 32);
    f8v s0 = zero, s1 = zero;
    s0 = wmma_f16(A0, b00, s0);
    s1 = wmma_f16(A0, b10, s1);
    s0 = wmma_f16(A1, b01, s0);
    s1 = wmma_f16(A1, b11, s1);
    if (!mk0) {
#pragma unroll
      for (int i = 0; i < 8; ++i) s0[i] = -1.0e30f;
    }
    if (!mk1) {
#pragma unroll
      for (int i = 0; i < 8; ++i) s1[i] = -1.0e30f;
    }

    // online softmax; rows live across 16 lanes of a half-wave
    float pr0[8], pr1[8];
#pragma unroll
    for (int i = 0; i < 8; ++i) {
      float t = fmaxf(s0[i], s1[i]);
      t = fmaxf(t, __shfl_xor(t, 1));
      t = fmaxf(t, __shfl_xor(t, 2));
      t = fmaxf(t, __shfl_xor(t, 4));
      t = fmaxf(t, __shfl_xor(t, 8));
      float mnew  = fmaxf(mrow[i], t);
      float alpha = __expf(mrow[i] - mnew);
      float p0 = __expf(s0[i] - mnew);
      float p1 = __expf(s1[i] - mnew);
      float rs = p0 + p1;
      rs += __shfl_xor(rs, 1);
      rs += __shfl_xor(rs, 2);
      rs += __shfl_xor(rs, 4);
      rs += __shfl_xor(rs, 8);
      lrow[i] = lrow[i] * alpha + rs;
      mrow[i] = mnew;
      pr0[i] = p0;
      pr1[i] = p1;
      o[0][i] *= alpha; o[1][i] *= alpha; o[2][i] *= alpha; o[3][i] *= alpha;
    }

    // transpose P (C-layout -> A-layout) through per-wave LDS slice
#pragma unroll
    for (int i = 0; i < 8; ++i) {
      lbuf[widx][(i + hi8) * 32 + n]      = (_Float16)pr0[i];
      lbuf[widx][(i + hi8) * 32 + 16 + n] = (_Float16)pr1[i];
    }
    __syncthreads();
    h16v Pa;
    {
      h8v lo = *(const h8v*)&lbuf[widx][n * 32 + dhalf];
      h8v hh = *(const h8v*)&lbuf[widx][n * 32 + dhalf + 8];
#pragma unroll
      for (int e = 0; e < 8; ++e) { Pa[e] = lo[e]; Pa[8 + e] = hh[e]; }
    }
    __syncthreads();

    // O += P @ V : issue all 4 V B-tile loads, then 4 WMMAs
    h16v Bv[4];
#pragma unroll
    for (int t = 0; t < 4; ++t)
      Bv[t] = *(const h16v*)(vp + (size_t)(t * 16 + n) * S_LEN + kb + dhalf);
#pragma unroll
    for (int t = 0; t < 4; ++t) o[t] = wmma_f16(Pa, Bv[t], o[t]);
  }

#pragma unroll
  for (int i = 0; i < 8; ++i) {
    float invl = 1.0f / fmaxf(lrow[i], 1e-20f);
    o[0][i] *= invl; o[1][i] *= invl; o[2][i] *= invl; o[3][i] *= invl;
  }

  // store O row-major f16 [token][h*64+d] via LDS transpose
#pragma unroll
  for (int t = 0; t < 4; ++t)
#pragma unroll
    for (int i = 0; i < 8; ++i)
      lbuf[widx][(i + hi8) * 64 + t * 16 + n] = (_Float16)o[t][i];
  __syncthreads();
  int colb = hi * 32;
  size_t token = (size_t)br * S_LEN + q0 + n;
  _Float16* dst = attnb + token * DM + h * HD + colb;
#pragma unroll
  for (int c = 0; c < 4; ++c) {
    h8v v = *(h8v*)&lbuf[widx][n * 64 + colb + c * 8];
    *(h8v*)(dst + c * 8) = v;
  }
}

// ---------------------------------------------------------------------------
// Kernel 3: output projection attn[16384x768]f16 @ Wo + bo -> f32 out.
// ---------------------------------------------------------------------------
__global__ void __launch_bounds__(128) k_proj(
    const _Float16* __restrict__ attnb, const _Float16* __restrict__ wsw,
    const float* __restrict__ bo, float* __restrict__ out) {
  __shared__ __align__(32) float fbuf[4][16 * 64];
  int widx = threadIdx.x >> 5;
  int lane = threadIdx.x & 31;
  int wv   = blockIdx.x * 4 + widx;
  int nt64 = wv >> 10;   // 0..11 (uniform per block)
  int rt   = wv & 1023;
  int n    = lane & 15;
  int hi   = (lane >> 4) & 1;
  int hi8  = hi * 8;

  const _Float16* arow = attnb + (size_t)(rt * 16 + n) * DM;
  const _Float16* wb   = wsw + (size_t)3 * WPERM
                       + (size_t)(nt64 * 4) * KCH * 512 + lane * 16;

  f8v zero = {};
  f8v acc[4];
#pragma unroll
  for (int t = 0; t < 4; ++t) acc[t] = zero;

  for (int kc = 0; kc < KCH; ++kc) {
    int kb0 = kc * 32;
    h16v B[4];
#pragma unroll
    for (int t = 0; t < 4; ++t)
      B[t] = *(const h16v*)(wb + (size_t)(t * KCH + kc) * 512);
    h8v lo = *(const h8v*)(arow + kb0 + hi8);
    h8v hh = *(const h8v*)(arow + kb0 + 16 + hi8);
    h16v A;
#pragma unroll
    for (int e = 0; e < 8; ++e) { A[e] = lo[e]; A[8 + e] = hh[e]; }
#pragma unroll
    for (int t = 0; t < 4; ++t) acc[t] = wmma_f16(A, B[t], acc[t]);
  }
#pragma unroll
  for (int t = 0; t < 4; ++t) {
    float bvv = bo[nt64 * 64 + t * 16 + n];
#pragma unroll
    for (int i = 0; i < 8; ++i) acc[t][i] += bvv;
  }
#pragma unroll
  for (int t = 0; t < 4; ++t)
#pragma unroll
    for (int i = 0; i < 8; ++i)
      fbuf[widx][(i + hi8) * 64 + t * 16 + n] = acc[t][i];
  __syncthreads();
  int colb = hi * 32;
  float* dst = out + (size_t)(rt * 16 + n) * DM + nt64 * 64 + colb;
#pragma unroll
  for (int c = 0; c < 8; ++c) {
    f4v v = *(f4v*)&fbuf[widx][n * 64 + colb + c * 4];
    *(f4v*)(dst + c * 4) = v;
  }
}

// ---------------------------------------------------------------------------
extern "C" void kernel_launch(void* const* d_in, const int* in_sizes, int n_in,
                              void* d_out, int out_size, void* d_ws, size_t ws_size,
                              hipStream_t stream) {
  (void)in_sizes; (void)n_in; (void)out_size; (void)ws_size;
  const float* x  = (const float*)d_in[0];
  const float* Wq = (const float*)d_in[1];
  const float* bq = (const float*)d_in[2];
  const float* Wk = (const float*)d_in[3];
  const float* bk = (const float*)d_in[4];
  const float* Wv = (const float*)d_in[5];
  const float* bv = (const float*)d_in[6];
  const float* Wo = (const float*)d_in[7];
  const float* bo = (const float*)d_in[8];
  const int* mask = (const int*)d_in[9];
  float* out = (float*)d_out;

  _Float16* wsh = (_Float16*)d_ws;
  _Float16* wsw = wsh;                         // 4 * 589824 halves (swizzled W)
  _Float16* qb  = wsw + (size_t)4 * WPERM;     // 192*1024*64 each below
  _Float16* kb  = qb + (size_t)192 * 1024 * 64;
  _Float16* vt  = kb + (size_t)192 * 1024 * 64;
  _Float16* ab  = vt + (size_t)192 * 1024 * 64;

  k_prep<<<9216, 256, 0, stream>>>(Wq, Wk, Wv, Wo, wsw);
  k_qkv <<<9216, 128, 0, stream>>>(x, bq, bk, bv, wsw, qb, kb, vt);
  k_attn<<<3072, 128, 0, stream>>>(qb, kb, vt, mask, ab);
  k_proj<<<3072, 128, 0, stream>>>(ab, wsw, bo, out);
}
